// GNN_node_expander_77687368450388
// MI455X (gfx1250) — matrix operand order repaired and checked by reading, not compiled, not run on hardware
//
#include <hip/hip_runtime.h>

typedef __attribute__((ext_vector_type(16))) __bf16 v16bf;
typedef __attribute__((ext_vector_type(8)))  __bf16 v8bf;
typedef __attribute__((ext_vector_type(8)))  float  v8f;
typedef unsigned int uint32x4 __attribute__((ext_vector_type(4)));
typedef int          int32x4  __attribute__((ext_vector_type(4)));
typedef int          int32x8  __attribute__((ext_vector_type(8)));

#if __has_builtin(__builtin_amdgcn_tensor_load_to_lds)
#define GEMM_USE_TDM 1
#else
#define GEMM_USE_TDM 0
#endif

__device__ __forceinline__ unsigned short f2bf(float f) {
  unsigned int u = __float_as_uint(f);
  if ((u & 0x7fffffffu) > 0x7f800000u) return (unsigned short)((u >> 16) | 0x0040u); // NaN
  unsigned int r = u + 0x7fffu + ((u >> 16) & 1u); // round-to-nearest-even
  return (unsigned short)(r >> 16);
}

// ---------------------------------------------------------------- utilities
__global__ void fill_zero(float* __restrict__ p, long n) {
  long i = (long)blockIdx.x * blockDim.x + threadIdx.x;
  long st = (long)gridDim.x * blockDim.x;
  for (; i < n; i += st) p[i] = 0.0f;
}

__global__ void f32_to_bf16_k(const float* __restrict__ s, unsigned short* __restrict__ d, long n) {
  long i = (long)blockIdx.x * blockDim.x + threadIdx.x;
  long st = (long)gridDim.x * blockDim.x;
  for (; i < n; i += st) d[i] = f2bf(s[i]);
}

// ---------------------------------------------------------------- encoders
// One block (64 threads) per node; each thread owns 4 floats of the E=256 row.
__global__ void atom_encoder(const int* __restrict__ x, const float* __restrict__ atom_emb,
                             float* __restrict__ h, int E) {
  int n = blockIdx.x;
  int t = threadIdx.x; // 0..63
  float4 acc = make_float4(0.f, 0.f, 0.f, 0.f);
#pragma unroll
  for (int f = 0; f < 9; ++f) {
    int idx = x[n * 9 + f];
    const float4* src = (const float4*)(atom_emb + ((size_t)f * 128 + idx) * E);
    float4 v = src[t];
    acc.x += v.x; acc.y += v.y; acc.z += v.z; acc.w += v.w;
  }
  ((float4*)(h + (size_t)n * E))[t] = acc;
}

// ---------------------------------------------------------------- edge scatter
// 4 edges per 256-thread block; 64 threads x float4 per edge row.
__global__ void edge_agg_bond(const float* __restrict__ h, const int* __restrict__ srcIdx,
                              const int* __restrict__ dstIdx, const int* __restrict__ eattr,
                              const float* __restrict__ bondE, float* __restrict__ agg,
                              int nE, int E) {
  int e = blockIdx.x * 4 + (threadIdx.x >> 6);
  if (e >= nE) return;
  int t = threadIdx.x & 63;
  int s = srcIdx[e], d = dstIdx[e];
  int a0 = eattr[3 * e + 0], a1 = eattr[3 * e + 1], a2 = eattr[3 * e + 2];
  float4 v  = ((const float4*)(h + (size_t)s * E))[t];
  float4 x0 = ((const float4*)(bondE + (size_t)(0 * 8 + a0) * E))[t];
  float4 x1 = ((const float4*)(bondE + (size_t)(1 * 8 + a1) * E))[t];
  float4 x2 = ((const float4*)(bondE + (size_t)(2 * 8 + a2) * E))[t];
  float m0 = fmaxf(v.x + x0.x + x1.x + x2.x, 0.f);
  float m1 = fmaxf(v.y + x0.y + x1.y + x2.y, 0.f);
  float m2 = fmaxf(v.z + x0.z + x1.z + x2.z, 0.f);
  float m3 = fmaxf(v.w + x0.w + x1.w + x2.w, 0.f);
  float* ap = agg + (size_t)d * E + t * 4;
  atomicAdd(ap + 0, m0); atomicAdd(ap + 1, m1);
  atomicAdd(ap + 2, m2); atomicAdd(ap + 3, m3);
}

__global__ void edge_agg_plain(const float* __restrict__ h, const int* __restrict__ srcIdx,
                               const int* __restrict__ dstIdx, float* __restrict__ agg,
                               int nE, int E) {
  int e = blockIdx.x * 4 + (threadIdx.x >> 6);
  if (e >= nE) return;
  int t = threadIdx.x & 63;
  int s = srcIdx[e], d = dstIdx[e];
  float4 v = ((const float4*)(h + (size_t)s * E))[t];
  float m0 = fmaxf(v.x, 0.f), m1 = fmaxf(v.y, 0.f);
  float m2 = fmaxf(v.z, 0.f), m3 = fmaxf(v.w, 0.f);
  float* ap = agg + (size_t)d * E + t * 4;
  atomicAdd(ap + 0, m0); atomicAdd(ap + 1, m1);
  atomicAdd(ap + 2, m2); atomicAdd(ap + 3, m3);
}

// h_in = (1+eps) * (mask? mask*h : h) + agg  -> bf16 GEMM operand
__global__ void combine(const float* __restrict__ h, const float* __restrict__ agg,
                        const unsigned char* __restrict__ mask, const float* __restrict__ epsPtr,
                        unsigned short* __restrict__ Abf, long total, int eShift) {
  long i = (long)blockIdx.x * blockDim.x + threadIdx.x;
  if (i >= total) return;
  float eps = epsPtr[0];
  float xs = h[i];
  if (mask) { long n = i >> eShift; xs *= mask[n] ? 1.0f : 0.0f; }
  Abf[i] = f2bf((1.0f + eps) * xs + agg[i]);
}

// ---------------------------------------------------------------- WMMA GEMM
// C[Mr,Nc] (f32) = A[Mr,K] (bf16) x B[K,Nc] (bf16). 128x128 block tile,
// 8 waves, each wave = 16-row band x 8 col tiles, K-step 32.
// A tile staged by the Tensor Data Mover (async DMA, TENSORcnt) with LDS
// padding so rows land at the bank-conflict-free stride; B tile staged
// manually (needs transpose). LDS double-buffered; B fragments pipelined
// with an explicit depth-2 rotation so >=2 ds_loads stay in flight.
#define TM 128
#define TN 128
#define TK 32
#define LSTR 40  // padded LDS row stride in bf16 elements (keeps 16B alignment)

__launch_bounds__(256)
__global__ void gemm_bf16_wmma(const unsigned short* __restrict__ Ag,
                               const unsigned short* __restrict__ Bg,
                               float* __restrict__ C, int Mr, int K, int Nc) {
  __shared__ __bf16 Alds[2 * TM * LSTR];   // [buf][row][k]
  __shared__ __bf16 Blds[2 * TN * LSTR];   // [buf][col][k]  (transposed)
  const int r0 = blockIdx.x * TM;
  const int c0 = blockIdx.y * TN;
  const int tid  = threadIdx.x;
  const int wave = tid >> 5;
  const int lane = tid & 31;
  const int lr = lane & 15;
  const int kh = lane >> 4;

  v8f acc[8];
#pragma unroll
  for (int c = 0; c < 8; ++c) acc[c] = v8f{};

  uint2 br[2][2];   // B staging: 2 groups x 2 K-rows x (4 bf16)

#if GEMM_USE_TDM
  // ---- Tensor Data Mover descriptor for the A tile (2-D, padded LDS dest).
  // Row = 32 bf16 = 16 DWORDs; pad_interval=16 DW (enc 3), pad_amount=4 DW
  // (enc 3) -> LDS stride 20 DW = 40 bf16 = LSTR.
  const unsigned ldsA0 = (unsigned)(size_t)(void*)&Alds[0];
  auto issueA = [&](int k0, int buf) {
    unsigned long long ga =
        (unsigned long long)(size_t)(const void*)(Ag + (size_t)r0 * K + k0);
    unsigned td0 = (unsigned)(K - k0);   // elements remaining along dim0
    unsigned td1 = (unsigned)(Mr - r0);  // rows remaining (OOB rows -> zeros)
    unsigned long long st0 = (unsigned long long)(unsigned)K;
    uint32x4 g0;
    g0.x = 1u;                                              // count=1, user mode
    g0.y = ldsA0 + (unsigned)(buf * TM * LSTR * 2);         // lds_addr (bytes)
    g0.z = (unsigned)ga;                                    // global_addr[31:0]
    g0.w = (unsigned)((ga >> 32) & 0x01FFFFFFu) | (2u << 30); // addr[56:32], type=2
    int32x8 g1;
    g1[0] = (int)((1u << 16) | (1u << 20) | (3u << 22) | (3u << 25)); // 2B, pad 16DW+4DW
    g1[1] = (int)((td0 & 0xFFFFu) << 16);                   // tensor_dim0[15:0]
    g1[2] = (int)(((td0 >> 16) & 0xFFFFu) | ((td1 & 0xFFFFu) << 16));
    g1[3] = (int)(((td1 >> 16) & 0xFFFFu) | ((unsigned)TK << 16)); // tile_dim0=32
    g1[4] = (int)TM;                                        // tile_dim1=128, tile_dim2=0
    g1[5] = (int)(unsigned)(st0 & 0xFFFFFFFFu);             // dim0 stride = K
    g1[6] = (int)(unsigned)((st0 >> 32) & 0xFFFFu);
    g1[7] = 0;
    int32x4 gz = {0, 0, 0, 0};
#if defined(__clang_major__) && __clang_major__ >= 23
    int32x8 gz8 = {0, 0, 0, 0, 0, 0, 0, 0};
    __builtin_amdgcn_tensor_load_to_lds(g0, g1, gz, gz, gz8, 0);
#else
    __builtin_amdgcn_tensor_load_to_lds(g0, g1, gz, gz, 0);
#endif
  };
#else
  uint2 ar[4];
  auto loadGA = [&](int k0) {
#pragma unroll
    for (int i = 0; i < 4; ++i) {
      int idx = (tid + i * 256) * 4;
      int row = idx >> 5, kk = idx & 31;
      int grow = r0 + row;
      if (grow >= Mr) grow = Mr - 1;
      ar[i] = *(const uint2*)(Ag + (size_t)grow * K + k0 + kk);
    }
  };
  auto storeLA = [&](int buf) {
    __bf16* Al = Alds + buf * TM * LSTR;
#pragma unroll
    for (int i = 0; i < 4; ++i) {
      int idx = (tid + i * 256) * 4;
      int row = idx >> 5, kk = idx & 31;
      *(uint2*)(&Al[row * LSTR + kk]) = ar[i];
    }
  };
#endif

  auto loadGB = [&](int k0) {
#pragma unroll
    for (int i = 0; i < 2; ++i) {
      int g = tid + i * 256;
      int q = g & 31, kp = g >> 5;                 // col-quad, K-pair
      const unsigned short* bp = Bg + (size_t)(k0 + kp * 2) * Nc + c0 + q * 4;
      br[i][0] = *(const uint2*)bp;
      br[i][1] = *(const uint2*)(bp + Nc);
    }
  };
  auto storeLB = [&](int buf) {
    __bf16* Bl = Blds + buf * TN * LSTR;
#pragma unroll
    for (int i = 0; i < 2; ++i) {
      int g = tid + i * 256;
      int q = g & 31, kk = (g >> 5) * 2;
      const unsigned short* p0 = (const unsigned short*)&br[i][0];
      const unsigned short* p1 = (const unsigned short*)&br[i][1];
#pragma unroll
      for (int j = 0; j < 4; ++j) {
        unsigned int packed = (unsigned int)p0[j] | ((unsigned int)p1[j] << 16);
        *(unsigned int*)(&Bl[(q * 4 + j) * LSTR + kk]) = packed;
      }
    }
  };

  auto loadBfrag = [&](const __bf16* Bl, int c) -> v16bf {
    int bcol = c * 16 + lr;
    v8bf blo = *(const v8bf*)(&Bl[bcol * LSTR + kh * 8]);
    v8bf bhi = *(const v8bf*)(&Bl[bcol * LSTR + 16 + kh * 8]);
    return __builtin_shufflevector(blo, bhi, 0,1,2,3,4,5,6,7,8,9,10,11,12,13,14,15);
  };

#if GEMM_USE_TDM
  if (tid < 32) issueA(0, 0);    // wave 0 owns the TDM queue for this block
#else
  loadGA(0);
  storeLA(0);
#endif
  loadGB(0);
  storeLB(0);

  const int nk = K / TK;
  for (int kt = 0; kt < nk; ++kt) {
#if GEMM_USE_TDM
    if (tid < 32) __builtin_amdgcn_s_wait_tensorcnt(0);  // tile kt landed in LDS
#endif
    __syncthreads();                       // staged tile visible to all waves
    const int buf = kt & 1;
    const bool more = (kt + 1 < nk);
    if (more) {
#if GEMM_USE_TDM
      if (tid < 32) issueA((kt + 1) * TK, 1 - buf);  // DMA next A tile
#else
      loadGA((kt + 1) * TK);
#endif
      loadGB((kt + 1) * TK);               // overlap global loads with compute
    }

    const __bf16* Al = Alds + buf * TM * LSTR;
    const __bf16* Bl = Blds + buf * TN * LSTR;
    // A fragment (16x32, ISA layout: lanes0-15 K{0..7,16..23}; lanes16-31 K{8..15,24..31})
    const int arow = wave * 16 + lr;
    v8bf alo = *(const v8bf*)(&Al[arow * LSTR + kh * 8]);
    v8bf ahi = *(const v8bf*)(&Al[arow * LSTR + 16 + kh * 8]);
    v16bf a = __builtin_shufflevector(alo, ahi, 0,1,2,3,4,5,6,7,8,9,10,11,12,13,14,15);

    // depth-2 pipelined B fragments: keep >=2 ds_load pairs in flight
    v16bf b0 = loadBfrag(Bl, 0);
    v16bf b1 = loadBfrag(Bl, 1);
#pragma unroll
    for (int c = 0; c < 8; ++c) {
      v16bf b2;
      if (c + 2 < 8) b2 = loadBfrag(Bl, c + 2);
      acc[c] = __builtin_amdgcn_wmma_f32_16x16x32_bf16(false, a, false, b0,
                                                       (short)0, acc[c], false, false);
      b0 = b1;
      b1 = b2;
    }

    if (more) {
      __syncthreads();                     // all waves done reading buf^1 from kt-1
#if !GEMM_USE_TDM
      storeLA(1 - buf);
#endif
      storeLB(1 - buf);
    }
  }

  // epilogue: lanes0-15 VGPR v -> M=v; lanes16-31 VGPR v -> M=8+v; N=lane&15
  const int rhalf = (lane < 16) ? 0 : 8;
#pragma unroll
  for (int c = 0; c < 8; ++c) {
    int col = c0 + c * 16 + lr;
#pragma unroll
    for (int v = 0; v < 8; ++v) {
      int row = r0 + wave * 16 + rhalf + v;
      if (row < Mr) C[(size_t)row * Nc + col] = acc[c][v];
    }
  }
}

// ---------------------------------------------------------------- batch-norm
__global__ void col_sums(const float* __restrict__ X, int N, int C, int rowsPerBlock,
                         float* __restrict__ sums) {
  int r0 = blockIdx.x * rowsPerBlock;
  int rend = r0 + rowsPerBlock; if (rend > N) rend = N;
  for (int cb = 0; cb < C; cb += blockDim.x) {
    int c = cb + threadIdx.x;
    if (c >= C) break;
    float s = 0.f, s2 = 0.f;
    for (int r = r0; r < rend; ++r) {
      float v = X[(size_t)r * C + c];
      s += v; s2 += v * v;
    }
    atomicAdd(&sums[c], s);
    atomicAdd(&sums[C + c], s2);
  }
}

__global__ void finalize_stats(const float* __restrict__ sums, int C, float invN,
                               float* __restrict__ mu, float* __restrict__ istd) {
  int c = blockIdx.x * blockDim.x + threadIdx.x;
  if (c >= C) return;
  float m = sums[c] * invN;
  float var = sums[C + c] * invN - m * m;
  mu[c] = m;
  istd[c] = rsqrtf(var + 1e-5f);
}

__global__ void bn_apply(const float* __restrict__ X, const float* __restrict__ mu,
                         const float* __restrict__ istd, const float* __restrict__ g,
                         const float* __restrict__ b, long total, int cMask, int doRelu,
                         float* __restrict__ outf, unsigned short* __restrict__ outbf) {
  long i = (long)blockIdx.x * blockDim.x + threadIdx.x;
  if (i >= total) return;
  int c = (int)(i & cMask);     // C is a power of two here
  float v = (X[i] - mu[c]) * istd[c] * g[c] + b[c];
  if (doRelu) v = fmaxf(v, 0.f);
  if (outf)  outf[i] = v;
  if (outbf) outbf[i] = f2bf(v);
}

// ---------------------------------------------------------------- launch
extern "C" void kernel_launch(void* const* d_in, const int* in_sizes, int n_in,
                              void* d_out, int out_size, void* d_ws, size_t ws_size,
                              hipStream_t stream) {
  (void)n_in; (void)out_size; (void)ws_size;
  const int* x    = (const int*)d_in[0];
  const int* ei   = (const int*)d_in[1];
  const int* ea   = (const int*)d_in[2];
  const int* eei  = (const int*)d_in[3];
  const unsigned char* mask = (const unsigned char*)d_in[4];
  const float* atom_emb = (const float*)d_in[5];
  const float* bond_emb = (const float*)d_in[6];
  const float* eps_m  = (const float*)d_in[7];
  const float* W1_m   = (const float*)d_in[8];
  // d_in[9]  b1_m: cancels inside BatchNorm (mean-subtracted) -> unused
  const float* g1_m   = (const float*)d_in[10];
  const float* be1_m  = (const float*)d_in[11];
  const float* W2_m   = (const float*)d_in[12];
  // d_in[13] b2_m: cancels inside outer BatchNorm -> unused
  const float* bn_g_m = (const float*)d_in[14];
  const float* bn_b_m = (const float*)d_in[15];
  const float* eps_e  = (const float*)d_in[16];
  const float* W1_e   = (const float*)d_in[17];
  const float* g1_e   = (const float*)d_in[18];
  const float* be1_e  = (const float*)d_in[19];
  const float* W2_e   = (const float*)d_in[20];
  const float* bn_g_e = (const float*)d_in[21];
  const float* bn_b_e = (const float*)d_in[22];

  const int L  = in_sizes[7];
  const int N  = in_sizes[0] / 9;
  const int M  = in_sizes[1] / 2;
  const int Mx = in_sizes[3] / 2;
  const int E  = in_sizes[5] / (9 * 128);
  const int H  = in_sizes[8] / (L * E);
  const int eShift = __builtin_ctz(E);

  // ---- workspace carve (~123 MB; fits in the 192 MB L2)
  size_t off = 0;
  auto carve = [&](size_t bytes) {
    char* p = (char*)d_ws + off;
    off += (bytes + 255) & ~(size_t)255;
    return (void*)p;
  };
  float* h             = (float*)carve((size_t)N * E * 4);
  float* agg           = (float*)carve((size_t)N * E * 4);
  float* t1            = (float*)carve((size_t)N * H * 4);
  float* t2            = (float*)carve((size_t)N * E * 4);
  unsigned short* Abf  = (unsigned short*)carve((size_t)N * H * 2);
  unsigned short* w1bf = (unsigned short*)carve((size_t)E * H * 2);
  unsigned short* w2bf = (unsigned short*)carve((size_t)H * E * 2);
  float* sums          = (float*)carve((size_t)2 * H * 4);
  float* mu            = (float*)carve((size_t)H * 4);
  float* istd          = (float*)carve((size_t)H * 4);

  auto stats_bn = [&](const float* X, int C, const float* g, const float* b,
                      int doRelu, float* outf, unsigned short* outbf) {
    fill_zero<<<8, 256, 0, stream>>>(sums, (long)2 * C);
    col_sums<<<(N + 255) / 256, 256, 0, stream>>>(X, N, C, 256, sums);
    finalize_stats<<<(C + 255) / 256, 256, 0, stream>>>(sums, C, 1.0f / (float)N, mu, istd);
    long tot = (long)N * C;
    bn_apply<<<(int)((tot + 255) / 256), 256, 0, stream>>>(X, mu, istd, g, b, tot, C - 1,
                                                           doRelu, outf, outbf);
  };

  auto run_conv = [&](const int* sIdx, const int* dIdx, int nEdges, bool withBond,
                      const float* epsPtr, bool useMask,
                      const float* W1, const float* g1, const float* be1,
                      const float* W2, const float* bg, const float* bb,
                      int finalRelu, float* outH) {
    long nAgg = (long)N * E;
    fill_zero<<<2048, 256, 0, stream>>>(agg, nAgg);
    if (withBond)
      edge_agg_bond<<<(nEdges + 3) / 4, 256, 0, stream>>>(h, sIdx, dIdx, ea, bond_emb,
                                                          agg, nEdges, E);
    else
      edge_agg_plain<<<(nEdges + 3) / 4, 256, 0, stream>>>(h, sIdx, dIdx, agg, nEdges, E);
    combine<<<(int)((nAgg + 255) / 256), 256, 0, stream>>>(
        h, agg, useMask ? mask : (const unsigned char*)nullptr, epsPtr, Abf, nAgg, eShift);
    f32_to_bf16_k<<<512, 256, 0, stream>>>(W1, w1bf, (long)E * H);
    {
      dim3 g((N + TM - 1) / TM, H / TN);
      gemm_bf16_wmma<<<g, 256, 0, stream>>>(Abf, w1bf, t1, N, E, H);
    }
    stats_bn(t1, H, g1, be1, 1, nullptr, Abf); // hidden BN+ReLU -> bf16 for GEMM2
    f32_to_bf16_k<<<512, 256, 0, stream>>>(W2, w2bf, (long)H * E);
    {
      dim3 g((N + TM - 1) / TM, E / TN);
      gemm_bf16_wmma<<<g, 256, 0, stream>>>(Abf, w2bf, t2, N, H, E);
    }
    stats_bn(t2, E, bg, bb, finalRelu, outH, nullptr); // outer BN (+ReLU)
  };

  // AtomEncoder
  atom_encoder<<<N, 64, 0, stream>>>(x, atom_emb, h, E);

  for (int l = 0; l < L; ++l) {
    // graph-edge conv (bond embeddings fused into the message kernel)
    run_conv(ei, ei + M, M, true, eps_m + l, false,
             W1_m + (size_t)l * E * H, g1_m + (size_t)l * H, be1_m + (size_t)l * H,
             W2_m + (size_t)l * H * E, bn_g_m + (size_t)l * E, bn_b_m + (size_t)l * E,
             1, h);
    // expander left conv (real -> expander), masked self term
    run_conv(eei, eei + Mx, Mx, false, eps_e + (size_t)l * 2 + 0, true,
             W1_e + (size_t)(l * 2 + 0) * E * H, g1_e + (size_t)(l * 2 + 0) * H,
             be1_e + (size_t)(l * 2 + 0) * H, W2_e + (size_t)(l * 2 + 0) * H * E,
             bn_g_e + (size_t)(l * 2 + 0) * E, bn_b_e + (size_t)(l * 2 + 0) * E,
             1, h);
    // expander right conv (reversed edges); last layer: no ReLU, write d_out
    bool last = (l == L - 1);
    run_conv(eei + Mx, eei, Mx, false, eps_e + (size_t)l * 2 + 1, true,
             W1_e + (size_t)(l * 2 + 1) * E * H, g1_e + (size_t)(l * 2 + 1) * H,
             be1_e + (size_t)(l * 2 + 1) * H, W2_e + (size_t)(l * 2 + 1) * H * E,
             bn_g_e + (size_t)(l * 2 + 1) * E, bn_b_e + (size_t)(l * 2 + 1) * E,
             last ? 0 : 1, last ? (float*)d_out : h);
  }
}
